// Decoder_8658654068997
// MI455X (gfx1250) — matrix operand order repaired and checked by reading, not compiled, not run on hardware
//
#include <hip/hip_runtime.h>
#include <hip/hip_bf16.h>

typedef __bf16 bf16;
typedef __attribute__((ext_vector_type(8)))  bf16  v8bf;
typedef __attribute__((ext_vector_type(16))) bf16  v16bf;
typedef __attribute__((ext_vector_type(8)))  float v8f;

// ---------------------------------------------------------------- helpers

__device__ __forceinline__ float sigf(float x) { return 1.0f / (1.0f + __expf(-x)); }

__device__ __forceinline__ bf16 f2bf(float f) {
    unsigned u = __float_as_uint(f);
    unsigned r = (u + 0x7FFFu + ((u >> 16) & 1u)) >> 16;   // round-to-nearest-even
    unsigned short h = (unsigned short)r;
    return __builtin_bit_cast(bf16, h);
}

__device__ __forceinline__ v8f wmma_bf16(v16bf a, v16bf b, v8f c) {
    return __builtin_amdgcn_wmma_f32_16x16x32_bf16(false, a, false, b, (short)0, c,
                                                   false, false);
}

#define SHUF16(lo, hi) __builtin_shufflevector(lo, hi, 0,1,2,3,4,5,6,7,8,9,10,11,12,13,14,15)

// ---------------------------------------------------------------- GEMM: C = A @ W^T (+bias)
// A: M x K bf16 row-major (lda), W: N x K bf16 row-major (ldb), C: M x N f32 (ldc).
// Block tile 128x64, 8 waves; each wave owns a 2x2 grid of 16x16 WMMA accumulators
// (32 rows x 32 cols) -> 4 WMMAs per 2 A-frags + 2 B-frags (2 b128 loads per WMMA).
__global__ void gemm_bf16_nt(const bf16* __restrict__ A, int lda,
                             const bf16* __restrict__ B, int ldb,
                             const float* __restrict__ bias,
                             float* __restrict__ C, long long ldc,
                             int M, int N, int K)
{
    const int lane = threadIdx.x & 31;
    const int wave = threadIdx.x >> 5;
    const int half = lane >> 4;
    const int l16  = lane & 15;
    const int m0 = blockIdx.y * 128 + (wave >> 1) * 32;
    const int n0 = blockIdx.x * 64  + (wave & 1) * 32;

    int mr0 = m0 + l16;      if (mr0 >= M) mr0 = M - 1;
    int mr1 = m0 + 16 + l16; if (mr1 >= M) mr1 = M - 1;
    int nc0 = n0 + l16;      if (nc0 >= N) nc0 = N - 1;
    int nc1 = n0 + 16 + l16; if (nc1 >= N) nc1 = N - 1;

    const bf16* pa0 = A + (long long)mr0 * lda + half * 16;
    const bf16* pa1 = A + (long long)mr1 * lda + half * 16;
    const bf16* pb0 = B + (long long)nc0 * ldb + half * 16;
    const bf16* pb1 = B + (long long)nc1 * ldb + half * 16;

    v8f acc00 = {}; v8f acc01 = {}; v8f acc10 = {}; v8f acc11 = {};
    for (int k = 0; k < K; k += 32) {
        v8bf a0lo = *(const v8bf*)(pa0 + k);
        v8bf a0hi = *(const v8bf*)(pa0 + k + 8);
        v8bf a1lo = *(const v8bf*)(pa1 + k);
        v8bf a1hi = *(const v8bf*)(pa1 + k + 8);
        v16bf af0 = SHUF16(a0lo, a0hi);
        v16bf af1 = SHUF16(a1lo, a1hi);
        v16bf b0 = *(const v16bf*)(pb0 + k);
        v16bf b1 = *(const v16bf*)(pb1 + k);
        acc00 = wmma_bf16(af0, b0, acc00);
        acc01 = wmma_bf16(af0, b1, acc01);
        acc10 = wmma_bf16(af1, b0, acc10);
        acc11 = wmma_bf16(af1, b1, acc11);
    }

    const int mb0 = m0 + 8 * half;
    const int mb1 = m0 + 16 + 8 * half;
    const int ncol0 = n0 + l16;
    const int ncol1 = n0 + 16 + l16;
    float bv0 = 0.0f, bv1 = 0.0f;
    if (bias) {
        if (ncol0 < N) bv0 = bias[ncol0];
        if (ncol1 < N) bv1 = bias[ncol1];
    }
#pragma unroll
    for (int r = 0; r < 8; ++r) {
        int ma = mb0 + r;
        if (ma < M) {
            if (ncol0 < N) C[(long long)ma * ldc + ncol0] = acc00[r] + bv0;
            if (ncol1 < N) C[(long long)ma * ldc + ncol1] = acc01[r] + bv1;
        }
        int mb = mb1 + r;
        if (mb < M) {
            if (ncol0 < N) C[(long long)mb * ldc + ncol0] = acc10[r] + bv0;
            if (ncol1 < N) C[(long long)mb * ldc + ncol1] = acc11[r] + bv1;
        }
    }
}

// ---------------------------------------------------------------- LSTM step (one of 32)
// grid 48 x 32 threads; block owns a 16-wide j-tile, all 4 gates in registers.
__global__ void lstm_step_kernel(const bf16* __restrict__ hprev_bf,  // 16 x 768
                                 bf16* __restrict__ hout_bf,         // 16 x 768
                                 const bf16* __restrict__ Whh,       // 3072 x 768
                                 const float* __restrict__ xg,       // 512 x 3072 (incl bias)
                                 float* __restrict__ hiddens,        // 512 x 768
                                 float* __restrict__ cells,          // 512 x 768
                                 int t)
{
    const int lane = threadIdx.x & 31;
    const int half = lane >> 4;
    const int l16  = lane & 15;
    const int j0   = blockIdx.x * 16;

    const bf16* pa = hprev_bf + l16 * 768 + half * 16;
    v8f acc[4] = {{}, {}, {}, {}};
    for (int k = 0; k < 768; k += 32) {
        v8bf alo = *(const v8bf*)(pa + k);
        v8bf ahi = *(const v8bf*)(pa + k + 8);
        v16bf af = SHUF16(alo, ahi);
#pragma unroll
        for (int g = 0; g < 4; ++g) {
            const bf16* pb = Whh + (long long)(g * 768 + j0 + l16) * 768 + k + half * 16;
            acc[g] = wmma_bf16(af, *(const v16bf*)pb, acc[g]);
        }
    }

    const int j = j0 + l16;
#pragma unroll
    for (int r = 0; r < 8; ++r) {
        int b = r + 8 * half;                       // batch row 0..15
        long long m = (long long)b * 32 + t;        // (b, t) row
        const float* xgr = xg + m * 3072;
        float gi = acc[0][r] + xgr[j];
        float gf = acc[1][r] + xgr[768 + j];
        float gg = acc[2][r] + xgr[1536 + j];
        float go = acc[3][r] + xgr[2304 + j];
        float cold = (t > 0) ? cells[(m - 1) * 768 + j] : 0.0f;
        float c = sigf(gf) * cold + sigf(gi) * tanhf(gg);
        float h = sigf(go) * tanhf(c);
        cells[m * 768 + j]   = c;
        hiddens[m * 768 + j] = h;
        hout_bf[b * 768 + j] = f2bf(h);
    }
}

// ---------------------------------------------------------------- elementwise kernels

__global__ void f32_to_bf16_kernel(const float* __restrict__ s, bf16* __restrict__ d,
                                   long long n)
{
    long long i = (long long)blockIdx.x * 256 + threadIdx.x;
    if (i < n) d[i] = f2bf(s[i]);
}

__global__ void zero_bf16_kernel(bf16* __restrict__ p, int n)
{
    int i = blockIdx.x * 256 + threadIdx.x;
    if (i < n) p[i] = f2bf(0.0f);
}

// x = [embed_W[captions], broadcast(v_g)] -> bf16, shape (512, 1536)
__global__ void build_x_kernel(const float* __restrict__ embW, const float* __restrict__ vg,
                               const int* __restrict__ caps, bf16* __restrict__ x_bf)
{
    long long i = (long long)blockIdx.x * 256 + threadIdx.x;
    if (i >= 512LL * 1536) return;
    int m = (int)(i / 1536), d = (int)(i % 1536);
    float v = (d < 768) ? embW[(long long)caps[m] * 768 + d]
                        : vg[(m / 32) * 768 + (d - 768)];
    x_bf[i] = f2bf(v);
}

// xmean[b,:] = mean_t x[b,t,:]  (16 x 1536) -> bf16
__global__ void mean_x_kernel(const float* __restrict__ embW, const float* __restrict__ vg,
                              const int* __restrict__ caps, bf16* __restrict__ xm_bf)
{
    int i = blockIdx.x * 256 + threadIdx.x;
    if (i >= 16 * 1536) return;
    int b = i / 1536, d = i % 1536;
    float s;
    if (d < 768) {
        s = 0.0f;
        for (int t = 0; t < 32; ++t) s += embW[(long long)caps[b * 32 + t] * 768 + d];
        s *= (1.0f / 32.0f);
    } else {
        s = vg[b * 768 + (d - 768)];
    }
    xm_bf[i] = f2bf(s);
}

// sentinel = sigmoid(sx + h_prev@sent_Wh^T) * tanh(cells)
__global__ void sentinel_kernel(const float* __restrict__ sx, const float* __restrict__ hh,
                                const float* __restrict__ cells,
                                float* __restrict__ sent, bf16* __restrict__ sent_bf)
{
    long long i = (long long)blockIdx.x * 256 + threadIdx.x;
    if (i >= 512LL * 768) return;
    int m = (int)(i / 768);
    int t = m & 31;
    float hp = (t > 0) ? hh[i - 768] : 0.0f;    // h_prev = hiddens[t-1] @ Wh
    float g = sigf(sx[i] + hp);
    float s = g * tanhf(cells[i]);
    sent[i] = s;
    sent_bf[i] = f2bf(s);
}

// switch_weights[b,k] = sigmoid( sum_j tanh(Vw+vgw+xmw) * sw_w[j] ) ; wave per (b,k)
__global__ void sw_weight_kernel(const float* __restrict__ Vw, const float* __restrict__ vgw,
                                 const float* __restrict__ xmw, const float* __restrict__ sww_w,
                                 float* __restrict__ sww)
{
    int wid = blockIdx.x * 8 + (threadIdx.x >> 5);
    int lane = threadIdx.x & 31;
    if (wid >= 784) return;
    int b = wid / 49;
    const float* pv = Vw  + (long long)wid * 768;
    const float* pg = vgw + b * 768;
    const float* pm = xmw + b * 768;
    float s = 0.0f;
    for (int j = lane; j < 768; j += 32)
        s += tanhf(pv[j] + pg[j] + pm[j]) * sww_w[j];
    for (int off = 16; off > 0; off >>= 1) s += __shfl_down(s, off, 32);
    if (lane == 0) sww[wid] = sigf(s);
}

__global__ void switch_out_kernel(const float* __restrict__ V, const float* __restrict__ sww,
                                  float* __restrict__ so, bf16* __restrict__ so_bf)
{
    long long i = (long long)blockIdx.x * 256 + threadIdx.x;
    if (i >= 784LL * 768) return;
    float v = sww[i / 768] * V[i];
    so[i] = v;
    so_bf[i] = f2bf(v);
}

// z[b,t,kk] ; kk<49 -> attention over regions, kk==49 -> sentinel score. wave per (b,t,kk)
__global__ void z_kernel(const float* __restrict__ cv, const float* __restrict__ ss,
                         const float* __restrict__ hg, const float* __restrict__ atw,
                         float* __restrict__ z)
{
    int wid = blockIdx.x * 8 + (threadIdx.x >> 5);
    int lane = threadIdx.x & 31;
    if (wid >= 512 * 50) return;
    int m  = wid / 50;           // b*32+t
    int kk = wid % 50;
    int b  = m >> 5;
    const float* p1 = (kk < 49) ? cv + (long long)(b * 49 + kk) * 768
                                : ss + (long long)m * 768;
    const float* p2 = hg + (long long)m * 768;
    float s = 0.0f;
    for (int j = lane; j < 768; j += 32)
        s += tanhf(p1[j] + p2[j]) * atw[j];
    for (int off = 16; off > 0; off >>= 1) s += __shfl_down(s, off, 32);
    if (lane == 0) z[m * 50 + kk] = s;
}

// softmax over 50, emit alpha/beta, c_hat+hiddens -> bf16. Block (128 thr) per (b,t)
__global__ void attend_kernel(const float* __restrict__ z, const float* __restrict__ so,
                              const float* __restrict__ sent, const float* __restrict__ hid,
                              float* __restrict__ out_alpha, float* __restrict__ out_beta,
                              bf16* __restrict__ chs_bf)
{
    __shared__ float sh[50];
    int m = blockIdx.x;
    int tid = threadIdx.x;
    if (tid < 50) sh[tid] = z[m * 50 + tid];
    __syncthreads();
    if (tid == 0) {
        float mx = sh[0];
        for (int k = 1; k < 50; ++k) mx = fmaxf(mx, sh[k]);
        float s = 0.0f;
        for (int k = 0; k < 50; ++k) { float e = __expf(sh[k] - mx); sh[k] = e; s += e; }
        float inv = 1.0f / s;
        for (int k = 0; k < 50; ++k) sh[k] *= inv;
    }
    __syncthreads();
    if (tid < 49) out_alpha[m * 49 + tid] = sh[tid];
    if (tid == 49) out_beta[m] = sh[49];
    float beta = sh[49];
    int b = m >> 5;
    const float* sob = so + (long long)b * 49 * 768;
    for (int j = tid; j < 768; j += 128) {
        float ct = 0.0f;
        for (int k = 0; k < 49; ++k) ct += sh[k] * sob[(long long)k * 768 + j];
        float chat = beta * sent[(long long)m * 768 + j] + (1.0f - beta) * ct;
        chs_bf[(long long)m * 768 + j] = f2bf(chat + hid[(long long)m * 768 + j]);
    }
}

// ---------------------------------------------------------------- host launcher

extern "C" void kernel_launch(void* const* d_in, const int* in_sizes, int n_in,
                              void* d_out, int out_size, void* d_ws, size_t ws_size,
                              hipStream_t stream)
{
    (void)in_sizes; (void)n_in; (void)out_size; (void)ws_size;

    const float* V_in   = (const float*)d_in[0];
    const float* v_g    = (const float*)d_in[1];
    const int*   caps   = (const int*)d_in[2];
    const float* embW   = (const float*)d_in[3];
    const float* Wih    = (const float*)d_in[4];
    const float* Whh    = (const float*)d_in[5];
    const float* lstm_b = (const float*)d_in[6];
    const float* sWx    = (const float*)d_in[7];
    const float* sWh    = (const float*)d_in[8];
    const float* swWg   = (const float*)d_in[9];
    const float* swWh   = (const float*)d_in[10];
    const float* swWx   = (const float*)d_in[11];
    const float* sw_w   = (const float*)d_in[12];
    const float* atWv   = (const float*)d_in[13];
    const float* atWg   = (const float*)d_in[14];
    const float* atWs   = (const float*)d_in[15];
    const float* at_w   = (const float*)d_in[16];
    const float* mlpW   = (const float*)d_in[17];
    const float* mlpb   = (const float*)d_in[18];

    const long long BT = 512, Hh = 768, E2 = 1536, G4 = 3072, VOC = 50257, BK = 784;

    char* wsb = (char*)d_ws;
    size_t off = 0;
    auto alloc = [&](long long bytes) -> void* {
        void* p = wsb + off;
        off += (size_t)((bytes + 255) & ~255LL);
        return p;
    };

    bf16*  x_bf    = (bf16*)alloc(BT * E2 * 2);
    bf16*  xm_bf   = (bf16*)alloc(16 * E2 * 2);
    bf16*  vg_bf   = (bf16*)alloc(16 * Hh * 2);
    bf16*  V_bf    = (bf16*)alloc(BK * Hh * 2);
    bf16*  wih_bf  = (bf16*)alloc(G4 * E2 * 2);
    bf16*  whh_bf  = (bf16*)alloc(G4 * Hh * 2);
    bf16*  wsx_bf  = (bf16*)alloc(Hh * E2 * 2);
    bf16*  wsh_bf  = (bf16*)alloc(Hh * Hh * 2);
    bf16*  wswg_bf = (bf16*)alloc(Hh * Hh * 2);
    bf16*  wswh_bf = (bf16*)alloc(Hh * Hh * 2);
    bf16*  wswx_bf = (bf16*)alloc(Hh * E2 * 2);
    bf16*  wav_bf  = (bf16*)alloc(Hh * Hh * 2);
    bf16*  wag_bf  = (bf16*)alloc(Hh * Hh * 2);
    bf16*  was_bf  = (bf16*)alloc(Hh * Hh * 2);
    bf16*  wmlp_bf = (bf16*)alloc(VOC * Hh * 2);
    float* xg      = (float*)alloc(BT * G4 * 4);
    float* hid     = (float*)alloc(BT * Hh * 4);
    float* cel     = (float*)alloc(BT * Hh * 4);
    bf16*  hid_bf  = (bf16*)alloc(BT * Hh * 2);
    bf16*  hbf     = (bf16*)alloc(33LL * 16 * Hh * 2);   // T+1 ping slots
    float* sx      = (float*)alloc(BT * Hh * 4);
    float* hh      = (float*)alloc(BT * Hh * 4);
    float* sent    = (float*)alloc(BT * Hh * 4);
    bf16*  sent_bf = (bf16*)alloc(BT * Hh * 2);
    float* Vw      = (float*)alloc(BK * Hh * 4);
    float* vgw     = (float*)alloc(16 * Hh * 4);
    float* xmw     = (float*)alloc(16 * Hh * 4);
    float* sww     = (float*)alloc(BK * 4);
    float* so      = (float*)alloc(BK * Hh * 4);
    bf16*  so_bf   = (bf16*)alloc(BK * Hh * 2);
    float* cvb     = (float*)alloc(BK * Hh * 4);
    float* hg      = (float*)alloc(BT * Hh * 4);
    float* ssb     = (float*)alloc(BT * Hh * 4);
    float* zb      = (float*)alloc(BT * 50 * 4);
    bf16*  chs_bf  = (bf16*)alloc(BT * Hh * 2);

    float* out_scores = (float*)d_out;
    float* out_alpha  = out_scores + BT * VOC;
    float* out_beta   = out_alpha + BT * 49;

    auto conv = [&](const float* s, bf16* d, long long n) {
        f32_to_bf16_kernel<<<(unsigned)((n + 255) / 256), 256, 0, stream>>>(s, d, n);
    };
    auto gemm = [&](const bf16* A, int lda, const bf16* B, int ldb, const float* bias,
                    float* C, long long ldc, int M, int N, int K) {
        dim3 g((N + 63) / 64, (M + 127) / 128);
        gemm_bf16_nt<<<g, 256, 0, stream>>>(A, lda, B, ldb, bias, C, ldc, M, N, K);
    };

    // 1) weight + input conversions to bf16
    conv(Wih,  wih_bf,  G4 * E2);
    conv(Whh,  whh_bf,  G4 * Hh);
    conv(sWx,  wsx_bf,  Hh * E2);
    conv(sWh,  wsh_bf,  Hh * Hh);
    conv(swWg, wswg_bf, Hh * Hh);
    conv(swWh, wswh_bf, Hh * Hh);
    conv(swWx, wswx_bf, Hh * E2);
    conv(atWv, wav_bf,  Hh * Hh);
    conv(atWg, wag_bf,  Hh * Hh);
    conv(atWs, was_bf,  Hh * Hh);
    conv(mlpW, wmlp_bf, VOC * Hh);
    conv(v_g,  vg_bf,   16 * Hh);
    conv(V_in, V_bf,    BK * Hh);

    // 2) build x, mean(x), zero h0
    build_x_kernel<<<(unsigned)((BT * E2 + 255) / 256), 256, 0, stream>>>(embW, v_g, caps, x_bf);
    mean_x_kernel<<<(16 * 1536 + 255) / 256, 256, 0, stream>>>(embW, v_g, caps, xm_bf);
    zero_bf16_kernel<<<(16 * 768 + 255) / 256, 256, 0, stream>>>(hbf, 16 * 768);

    // 3) input-gate GEMM (includes bias) and sentinel-x GEMM
    gemm(x_bf, (int)E2, wih_bf, (int)E2, lstm_b, xg, G4, (int)BT, (int)G4, (int)E2);
    gemm(x_bf, (int)E2, wsx_bf, (int)E2, nullptr, sx, Hh, (int)BT, (int)Hh, (int)E2);

    // 4) LSTM scan: 32 sequential steps
    for (int t = 0; t < 32; ++t) {
        lstm_step_kernel<<<48, 32, 0, stream>>>(hbf + (long long)t * 16 * 768,
                                                hbf + (long long)(t + 1) * 16 * 768,
                                                whh_bf, xg, hid, cel, t);
    }
    conv(hid, hid_bf, BT * Hh);

    // 5) sentinel path
    gemm(hid_bf, (int)Hh, wsh_bf, (int)Hh, nullptr, hh, Hh, (int)BT, (int)Hh, (int)Hh);
    sentinel_kernel<<<(unsigned)((BT * Hh + 255) / 256), 256, 0, stream>>>(sx, hh, cel, sent, sent_bf);

    // 6) switch path
    gemm(V_bf,  (int)Hh, wswg_bf, (int)Hh, nullptr, Vw,  Hh, (int)BK, (int)Hh, (int)Hh);
    gemm(vg_bf, (int)Hh, wswh_bf, (int)Hh, nullptr, vgw, Hh, 16, (int)Hh, (int)Hh);
    gemm(xm_bf, (int)E2, wswx_bf, (int)E2, nullptr, xmw, Hh, 16, (int)Hh, (int)E2);
    sw_weight_kernel<<<(784 + 7) / 8, 256, 0, stream>>>(Vw, vgw, xmw, sw_w, sww);
    switch_out_kernel<<<(unsigned)((BK * Hh + 255) / 256), 256, 0, stream>>>(V_in, sww, so, so_bf);

    // 7) attention projections
    gemm(so_bf,   (int)Hh, wav_bf, (int)Hh, nullptr, cvb, Hh, (int)BK, (int)Hh, (int)Hh);
    gemm(hid_bf,  (int)Hh, wag_bf, (int)Hh, nullptr, hg,  Hh, (int)BT, (int)Hh, (int)Hh);
    gemm(sent_bf, (int)Hh, was_bf, (int)Hh, nullptr, ssb, Hh, (int)BT, (int)Hh, (int)Hh);

    // 8) scores z, softmax, context, c_hat + h
    z_kernel<<<(512 * 50 + 7) / 8, 256, 0, stream>>>(cvb, ssb, hg, at_w, zb);
    attend_kernel<<<512, 128, 0, stream>>>(zb, so, sent, hid, out_alpha, out_beta, chs_bf);

    // 9) vocab projection -> d_out (dominant GEMM: 512 x 50257 x 768)
    gemm(chs_bf, (int)Hh, wmlp_bf, (int)Hh, mlpb, out_scores, VOC, (int)BT, (int)VOC, (int)Hh);
}
